// MPCP_23175643529318
// MI455X (gfx1250) — compile-verified, matrix operand order checked by reading, not run on hardware
//
#include <hip/hip_runtime.h>
#include <hip/hip_bf16.h>
#include <math.h>

// ---------------- model constants ----------------
constexpr int N  = 4000;
constexpr int E  = 48000;
constexpr int C  = 64;
constexpr int Z  = 10;
constexpr int NB = 8;
constexpr int NG = 8;
constexpr int NL = 2;
constexpr int M  = 16;          // spherical components
constexpr float RMAX = 5.0f;
constexpr float INV_AVG = 1.0f / 12.0f;

constexpr int NTILE = (N/16 + 3) / 4;   // 63 row-tile groups for node GEMMs
constexpr int ETILE = (E/16) / 4;       // 750 row-tile groups for edge GEMMs

#define DEV __device__ __forceinline__

typedef __attribute__((ext_vector_type(16))) _Float16 v16h;
typedef __attribute__((ext_vector_type(8)))  float    v8f;
typedef __attribute__((ext_vector_type(4)))  unsigned int u32x4;
typedef __attribute__((ext_vector_type(8)))  int      i32x8;
typedef __attribute__((ext_vector_type(4)))  int      i32x4;

DEV float sigmf(float x){ return 1.0f/(1.0f+expf(-x)); }
DEV float siluf(float x){ return x*sigmf(x); }
DEV float dsiluf(float x){ float s=sigmf(x); return s*(1.0f+x*(1.0f-s)); }
DEV int ell_of(int m){ return (m==0)?0:((m<4)?1:((m<9)?2:3)); }

DEV v8f wmma_step(v16h a, v16h b, v8f c){
    return __builtin_amdgcn_wmma_f32_16x16x32_f16(false, a, false, b,
                                                  (short)0, c, false, false);
}

// ---------------- Tensor Data Mover staging ----------------
#if defined(__has_builtin)
#if __has_builtin(__builtin_amdgcn_tensor_load_to_lds)
#define HAVE_TDM 1
#endif
#endif
#ifndef HAVE_TDM
#define HAVE_TDM 0
#endif

#if HAVE_TDM
// Stage a 2D f16 tile (h rows x w elems, row stride = stride elems) into LDS.
// D# built per CDNA5 ISA §8: group0 {count=1, lds_addr, global_addr, type=2},
// group1 {data_size=2B, tensor_dim0/1, tile_dim0/1, dim0_stride}.
DEV void tdm_load_2d(const void* g, unsigned lds, unsigned w, unsigned h, unsigned stride)
{
    unsigned long long ga = (unsigned long long)g;
    u32x4 g0 = { 1u, lds,
                 (unsigned)(ga & 0xffffffffu),
                 (unsigned)(((ga >> 32) & 0x1ffffffu) | (2u << 30)) };
    i32x8 g1 = { (int)(1u << 16),                              // data_size = 2 bytes
                 (int)((w & 0xffffu) << 16),                   // tensor_dim0 lo16
                 (int)(((w >> 16) & 0xffffu) | ((h & 0xffffu) << 16)), // dim0 hi | dim1 lo
                 (int)(((h >> 16) & 0xffffu) | ((w & 0xffffu) << 16)), // dim1 hi | tile_dim0
                 (int)(h & 0xffffu),                           // tile_dim1 (tile_dim2 = 0)
                 (int)stride,                                  // dim0_stride lo32
                 0, 0 };
    i32x4 gz = {0,0,0,0};
#if __clang_major__ >= 23
    i32x8 gz8 = {0,0,0,0,0,0,0,0};
    __builtin_amdgcn_tensor_load_to_lds(g0, g1, gz, gz, gz8, 0);
#else
    __builtin_amdgcn_tensor_load_to_lds(g0, g1, gz, gz, 0);
#endif
}
#endif

// ---------------- GEMM: D[Mr x Nc] = A[Mr x K] * B, B passed transposed ----------------
// block = 128 (4 waves), grid.x = ceil(Mr/64), grid.y = Nc/16.
// B panel (16 x K) staged to LDS (TDM if available), shared by the 4 waves;
// each wave computes one 16x16 tile with A kept in registers.
__global__ void gemm_k(const _Float16* __restrict__ A, int lda,
                       const _Float16* __restrict__ Bt, int ldb,
                       float* __restrict__ D, int ldd,
                       int K, int Mr, int accum)
{
    __shared__ _Float16 Bsh[16*256];
    const _Float16* Bg = Bt + (size_t)blockIdx.y * 16 * ldb;
#if HAVE_TDM
    if (threadIdx.x < 32) {
        tdm_load_2d(Bg, (unsigned)(size_t)Bsh, (unsigned)K, 16u, (unsigned)ldb);
        __builtin_amdgcn_s_wait_tensorcnt(0);
    }
#else
    for (int i = threadIdx.x; i < 16*K; i += blockDim.x)
        Bsh[(i / K) * K + (i % K)] = Bg[(size_t)(i / K) * ldb + (i % K)];
#endif
    __syncthreads();
    int wave = threadIdx.x >> 5, lane = threadIdx.x & 31;
    int rowTile = blockIdx.x * 4 + wave;
    if (rowTile * 16 >= Mr) return;          // wave-uniform; after barrier
    int lr = lane & 15, kh = (lane >> 4) << 4;
    const _Float16* Ap = A + (size_t)rowTile * 16 * lda + (size_t)lr * lda + kh;
    const _Float16* Bp = Bsh + lr * K + kh;
    v8f acc = {};
    for (int k0 = 0; k0 < K; k0 += 32) {
        v16h a = *(const v16h*)(Ap + k0);
        v16h b = *(const v16h*)(Bp + k0);
        acc = wmma_step(a, b, acc);
    }
    int col = lane & 15, rbase = (lane >> 4) * 8;
    float* Dp = D + (size_t)rowTile * 16 * ldd + (size_t)blockIdx.y * 16 + col;
#pragma unroll
    for (int r = 0; r < 8; ++r) {
        size_t off = (size_t)(rbase + r) * ldd;
        if (accum) Dp[off] += acc[r]; else Dp[off] = acc[r];
    }
}

// ---------------- batched per-l channel mixer ----------------
// D[n,m,d] (+)= sum_c A[n,m,c] * W[l(m)][c or d][..]; A,D: (N,16,64), W: (4,64,64) Bt-form.
// block = 128 (4 waves), grid.x = ceil(250/4), grid.y = 16 (m).
// Whole 64x64 weight for l(m) staged to LDS once; each wave holds its A tile in
// registers and sweeps all four 16-wide d-tiles -> 8 WMMA per wave per block.
__global__ void perl_k(const _Float16* __restrict__ A,
                       const _Float16* __restrict__ W,
                       float* __restrict__ D, int accum)
{
    __shared__ _Float16 Bsh[64*64];
    int m = blockIdx.y;
    int l = ell_of(m);
    const _Float16* Wg = W + (size_t)l * C * C;
#if HAVE_TDM
    if (threadIdx.x < 32) {
        tdm_load_2d(Wg, (unsigned)(size_t)Bsh, (unsigned)(C*C), 1u, (unsigned)(C*C));
        __builtin_amdgcn_s_wait_tensorcnt(0);
    }
#else
    for (int i = threadIdx.x; i < C*C; i += blockDim.x) Bsh[i] = Wg[i];
#endif
    __syncthreads();
    int wave = threadIdx.x >> 5, lane = threadIdx.x & 31;
    int rowTile = blockIdx.x * 4 + wave;
    if (rowTile >= N/16) return;             // wave-uniform; after barrier
    int lr = lane & 15, kh = (lane >> 4) << 4;
    const _Float16* Ap = A + ((size_t)rowTile * 16 + lr) * (M*C) + m * C + kh;
    v16h a0 = *(const v16h*)(Ap);
    v16h a1 = *(const v16h*)(Ap + 32);
    int col = lane & 15, rbase = (lane >> 4) * 8;
    float* Dp = D + (size_t)rowTile * 16 * (M*C) + m * C + col;
#pragma unroll
    for (int dt = 0; dt < 4; ++dt) {
        const _Float16* Bp = Bsh + (dt * 16 + lr) * C + kh;
        v16h b0 = *(const v16h*)(Bp);
        v16h b1 = *(const v16h*)(Bp + 32);
        v8f acc = {};
        acc = wmma_step(a0, b0, acc);
        acc = wmma_step(a1, b1, acc);
        float* Dq = Dp + dt * 16;
#pragma unroll
        for (int r = 0; r < 8; ++r) {
            size_t off = (size_t)(rbase + r) * (M*C);
            if (accum) Dq[off] += acc[r]; else Dq[off] = acc[r];
        }
    }
}

// ---------------- utility kernels ----------------
__global__ void zero_k(float* p, int n){
    int i = blockIdx.x*blockDim.x + threadIdx.x;
    if (i < n) p[i] = 0.0f;
}
__global__ void cvt_k(const float* __restrict__ s, _Float16* __restrict__ d, int n){
    int i = blockIdx.x*blockDim.x + threadIdx.x;
    if (i < n) d[i] = (_Float16)s[i];
}
__global__ void cvt_silu_k(const float* __restrict__ s, _Float16* __restrict__ d, int n){
    int i = blockIdx.x*blockDim.x + threadIdx.x;
    if (i < n) d[i] = (_Float16)siluf(s[i]);
}
// batched transpose+convert: dst[b][c][r] = src[b][r][c]
__global__ void bct_k(const float* __restrict__ s, _Float16* __restrict__ d,
                      int B, int R, int Cc){
    int i = blockIdx.x*blockDim.x + threadIdx.x;
    int tot = B*R*Cc;
    if (i >= tot) return;
    int b = i/(R*Cc); int rem = i%(R*Cc); int r = rem/Cc; int c = rem%Cc;
    d[(size_t)b*R*Cc + (size_t)c*R + r] = (_Float16)s[i];
}

// ---------------- edge geometry (fwd) ----------------
__global__ void edge_geom_k(const float* __restrict__ pos,
                            const float* __restrict__ shifts,
                            const int* __restrict__ ei,
                            float* __restrict__ vec, float* __restrict__ rlen,
                            float* __restrict__ ea, float* __restrict__ ef)
{
    int e = blockIdx.x*blockDim.x + threadIdx.x;
    if (e >= E) return;
    int s = ei[e], r = ei[E + e];
    float vx = pos[r*3+0]-pos[s*3+0]+shifts[e*3+0];
    float vy = pos[r*3+1]-pos[s*3+1]+shifts[e*3+1];
    float vz = pos[r*3+2]-pos[s*3+2]+shifts[e*3+2];
    float rr = sqrtf(vx*vx+vy*vy+vz*vz);
    vec[e*3+0]=vx; vec[e*3+1]=vy; vec[e*3+2]=vz; rlen[e]=rr;
    float inv = 1.0f/rr;
    float x=vx*inv, y=vy*inv, z=vz*inv;
    const float s3=1.7320508075688772f, s5=2.23606797749979f, s15=3.872983346207417f;
    const float s7=2.6457513110645907f, s105=10.246950765959598f;
    const float s35_8=2.091650066335189f, s21_8=1.6201851746019651f;
    float x2=x*x, y2=y*y, z2=z*z;
    float* a = ea + (size_t)e*16;
    a[0]=1.0f;  a[1]=s3*x;  a[2]=s3*y;  a[3]=s3*z;
    a[4]=s15*x*y; a[5]=s15*y*z; a[6]=0.5f*s5*(3.0f*z2-1.0f);
    a[7]=s15*x*z; a[8]=0.5f*s15*(x2-y2);
    a[9]=s35_8*y*(3.0f*x2-y2); a[10]=s105*x*y*z;
    a[11]=s21_8*y*(5.0f*z2-1.0f); a[12]=0.5f*s7*z*(5.0f*z2-3.0f);
    a[13]=s21_8*x*(5.0f*z2-1.0f); a[14]=0.5f*s105*z*(x2-y2);
    a[15]=s35_8*x*(x2-3.0f*y2);
    // radial
    float u = rr / RMAX;
    float u2=u*u, u4=u2*u2, u5=u4*u, u6=u5*u, u7=u6*u;
    float env = (u < 1.0f) ? (1.0f - 21.0f*u5 + 35.0f*u6 - 15.0f*u7) : 0.0f;
    const float c0 = 0.6324555320336759f; // sqrt(2/5)
    float* f = ef + (size_t)e*NB;
#pragma unroll
    for (int nb = 1; nb <= NB; ++nb) {
        float w = (float)nb * (float)M_PI / RMAX;
        f[nb-1] = c0 * sinf(w*rr) * inv * env;
    }
}

// ---------------- embedding ----------------
__global__ void embed_k(const float* __restrict__ x, const float* __restrict__ Wemb,
                        float* __restrict__ nf0){
    int i = blockIdx.x*blockDim.x + threadIdx.x;
    if (i >= N*M*C) return;
    int n = i >> 10; int m = (i >> 6) & 15; int c = i & 63;
    float v = 0.0f;
    if (m == 0) {
#pragma unroll
        for (int z = 0; z < Z; ++z) v += x[n*Z+z]*Wemb[z*C+c];
    }
    nf0[i] = v;
}
__global__ void asc_k(const float* __restrict__ x, const float* __restrict__ a_sc,
                      float* __restrict__ asc0, float* __restrict__ asc1){
    int i = blockIdx.x*blockDim.x + threadIdx.x;
    if (i >= N*C) return;
    int n = i >> 6; int c = i & 63;
    float v0=0.0f, v1=0.0f;
#pragma unroll
    for (int z = 0; z < Z; ++z) {
        float xv = x[n*Z+z];
        v0 += xv*a_sc[z*C+c];
        v1 += xv*a_sc[Z*C + z*C+c];
    }
    asc0[i]=v0; asc1[i]=v1;
}

// edge MLP layer 1 (K=8, VALU)
__global__ void mlp1_k(const float* __restrict__ ef, const float* __restrict__ Wm1,
                       float* __restrict__ z1){
    int i = blockIdx.x*blockDim.x + threadIdx.x;
    if (i >= E*C) return;
    int e = i >> 6; int h = i & 63;
    const float* f = ef + (size_t)e*NB;
    float v = 0.0f;
#pragma unroll
    for (int b = 0; b < NB; ++b) v += f[b]*Wm1[b*C+h];
    z1[i] = v;
}

// fused message + segment-sum: agg[r][m][c] += w3[e][c][l(m)] * ea[e][m] * h0[s][c] / 12
__global__ void msg_k(const float* __restrict__ w3, const float* __restrict__ ea,
                      const float* __restrict__ h0, const int* __restrict__ ei,
                      float* __restrict__ agg){
    int e = blockIdx.x; int c = threadIdx.x;
    int s = ei[e], r = ei[E+e];
    float hs = h0[(size_t)s*C + c];
    const float* w = w3 + (size_t)e*(4*C) + c*4;
    const float* ap = ea + (size_t)e*16;
    float* ag = agg + ((size_t)r << 10) + c;
    float w0=w[0], w1=w[1], w2=w[2], w3v=w[3];
#pragma unroll
    for (int m = 0; m < 16; ++m) {
        int l = ell_of(m);
        float wv = (l==0)?w0:((l==1)?w1:((l==2)?w2:w3v));
        atomicAdd(ag + m*C, wv*ap[m]*hs*INV_AVG);
    }
}

// gate + out*gate (as f16) ; also stores gate
__global__ void gate_og_k(const float* __restrict__ out, const float* __restrict__ wp,
                          float* __restrict__ gate, _Float16* __restrict__ og16){
    int i = blockIdx.x*blockDim.x + threadIdx.x;
    if (i >= N*M*C) return;
    int n = i >> 10; int m = (i >> 6) & 15; int c = i & 63;
    float s = out[((size_t)n << 10) + c];   // m = 0 component
    float g = wp[c] + wp[C+c]*s + wp[2*C+c]*s*s;
    if (m == 0) gate[(n<<6)+c] = g;
    og16[i] = (_Float16)(out[i]*g);
}

__global__ void scale_asc_k(float* __restrict__ nf, const float* __restrict__ asc){
    int i = blockIdx.x*blockDim.x + threadIdx.x;
    if (i >= N*M*C) return;
    int n = i >> 10; int c = i & 63;
    nf[i] *= asc[(n<<6)+c];
}

// readout
__global__ void ne0_k(const float* __restrict__ nf1, const float* __restrict__ w_r0,
                      float* __restrict__ node_e){
    int n = blockIdx.x*blockDim.x + threadIdx.x;
    if (n >= N) return;
    const float* p = nf1 + ((size_t)n << 10);
    float v = 0.0f;
    for (int c = 0; c < C; ++c) v += p[c]*w_r0[c];
    node_e[n] = v;
}
__global__ void ne1_k(const float* __restrict__ nf2, const float* __restrict__ W_r1,
                      const float* __restrict__ w_r2, float* __restrict__ node_e){
    int n = blockIdx.x*blockDim.x + threadIdx.x;
    if (n >= N) return;
    const float* p = nf2 + ((size_t)n << 10);
    float v = 0.0f;
    for (int k = 0; k < 16; ++k) {
        float t = 0.0f;
        for (int c = 0; c < C; ++c) t += p[c]*W_r1[c*16+k];
        v += siluf(t)*w_r2[k];
    }
    node_e[n] += v;
}
__global__ void snap_k(const float* __restrict__ node_e, const int* __restrict__ batch,
                       float* __restrict__ snap){
    int n = blockIdx.x*blockDim.x + threadIdx.x;
    if (n >= N) return;
    atomicAdd(&snap[batch[n]], node_e[n]);
}

// ---------------- backward kernels ----------------
__global__ void gnf2_init_k(const float* __restrict__ nf2, const float* __restrict__ W_r1,
                            const float* __restrict__ w_r2, float* __restrict__ g_nf2){
    int n = blockIdx.x*blockDim.x + threadIdx.x;
    if (n >= N) return;
    const float* p = nf2 + ((size_t)n << 10);
    float t[16];
    for (int k = 0; k < 16; ++k) {
        float tt = 0.0f;
        for (int c = 0; c < C; ++c) tt += p[c]*W_r1[c*16+k];
        t[k] = w_r2[k]*dsiluf(tt);
    }
    float* g = g_nf2 + ((size_t)n << 10);
    for (int c = 0; c < C; ++c) {
        float v = 0.0f;
        for (int k = 0; k < 16; ++k) v += t[k]*W_r1[c*16+k];
        g[c] = v;
    }
}

// g_og -> g_out (in place) + gate polynomial backward
__global__ void gate_bwd_k(const float* __restrict__ out, const float* __restrict__ gate,
                           const float* __restrict__ wp, float* __restrict__ g_og){
    int i = blockIdx.x*blockDim.x + threadIdx.x;
    if (i >= N*C) return;
    int n = i >> 6; int c = i & 63;
    float gt = gate[i];
    float* go = g_og + ((size_t)n << 10) + c;
    const float* op = out + ((size_t)n << 10) + c;
    float gg = 0.0f;
#pragma unroll
    for (int m = 0; m < 16; ++m) {
        float g0 = go[m*C];
        gg += g0*op[m*C];
        go[m*C] = g0*gt;
    }
    float s = op[0];
    go[0] += gg*(wp[C+c] + 2.0f*wp[2*C+c]*s);
}

// (g_nf * asc) -> f16
__global__ void gsc_cvt_k(const float* __restrict__ g_nf, const float* __restrict__ asc,
                          _Float16* __restrict__ d16){
    int i = blockIdx.x*blockDim.x + threadIdx.x;
    if (i >= N*M*C) return;
    int n = i >> 10; int c = i & 63;
    d16[i] = (_Float16)(g_nf[i]*asc[(n<<6)+c]);
}

__global__ void msg_bwd_k(const float* __restrict__ g_agg, const float* __restrict__ w3,
                          const float* __restrict__ ea, const float* __restrict__ h0,
                          const int* __restrict__ ei,
                          float* __restrict__ g_w3, float* __restrict__ g_h0,
                          float* __restrict__ g_ea){
    __shared__ float sh[16];
    int e = blockIdx.x; int c = threadIdx.x;
    if (c < 16) sh[c] = 0.0f;
    __syncthreads();
    int s = ei[e], r = ei[E+e];
    float hs = h0[(size_t)s*C + c];
    const float* ga = g_agg + ((size_t)r << 10) + c;
    const float* w  = w3 + (size_t)e*(4*C) + c*4;
    const float* ap = ea + (size_t)e*16;
    float gw[4] = {0,0,0,0};
    float gh = 0.0f;
#pragma unroll
    for (int m = 0; m < 16; ++m) {
        int l = ell_of(m);
        float gm = ga[m*C]*INV_AVG;
        float eav = ap[m];
        float wv = w[l];
        gw[l] += gm*eav*hs;
        gh    += gm*wv*eav;
        atomicAdd(&sh[m], gm*wv*hs);
    }
    float* gwp = g_w3 + (size_t)e*(4*C) + c*4;
    gwp[0]=gw[0]; gwp[1]=gw[1]; gwp[2]=gw[2]; gwp[3]=gw[3];
    atomicAdd(&g_h0[(size_t)s*C + c], gh);
    __syncthreads();
    if (c < 16) atomicAdd(&g_ea[(size_t)e*16 + c], sh[c]);
}

__global__ void dsilu_mul_k(float* __restrict__ g, const float* __restrict__ z, int n){
    int i = blockIdx.x*blockDim.x + threadIdx.x;
    if (i < n) g[i] *= dsiluf(z[i]);
}

__global__ void mlp1_bwd_k(const float* __restrict__ g_z1, const float* __restrict__ Wm1,
                           float* __restrict__ g_ef){
    int e = blockIdx.x*blockDim.x + threadIdx.x;
    if (e >= E) return;
    const float* g = g_z1 + (size_t)e*C;
#pragma unroll
    for (int b = 0; b < NB; ++b) {
        float v = 0.0f;
        for (int h = 0; h < C; ++h) v += g[h]*Wm1[b*C+h];
        g_ef[(size_t)e*NB + b] += v;
    }
}

__global__ void add_wr0_k(float* __restrict__ g_nf1, const float* __restrict__ w_r0){
    int i = blockIdx.x*blockDim.x + threadIdx.x;
    if (i >= N*C) return;
    int n = i >> 6; int c = i & 63;
    g_nf1[((size_t)n << 10) + c] += w_r0[c];
}

// geometric backward: vec-grad from g_ea (sph Jacobian) and g_ef (radial derivative)
__global__ void edge_bwd_k(const float* __restrict__ vec, const float* __restrict__ rlen,
                           const float* __restrict__ g_ea, const float* __restrict__ g_ef,
                           const int* __restrict__ ei, float* __restrict__ forces){
    int e = blockIdx.x*blockDim.x + threadIdx.x;
    if (e >= E) return;
    float r = rlen[e];
    float inv = 1.0f/r;
    float x = vec[e*3+0]*inv, y = vec[e*3+1]*inv, z = vec[e*3+2]*inv;
    const float s3=1.7320508075688772f, s5=2.23606797749979f, s15=3.872983346207417f;
    const float s7=2.6457513110645907f, s105=10.246950765959598f;
    const float s35_8=2.091650066335189f, s21_8=1.6201851746019651f;
    float x2=x*x, y2=y*y, z2=z*z;
    const float* g = g_ea + (size_t)e*16;
    float gx=0.0f, gy=0.0f, gz=0.0f;
    gx += g[1]*s3;                           gy += g[2]*s3;                 gz += g[3]*s3;
    gx += g[4]*s15*y;                        gy += g[4]*s15*x;
    gy += g[5]*s15*z;                        gz += g[5]*s15*y;
    gz += g[6]*3.0f*s5*z;
    gx += g[7]*s15*z;                        gz += g[7]*s15*x;
    gx += g[8]*s15*x;                        gy -= g[8]*s15*y;
    gx += g[9]*6.0f*s35_8*x*y;               gy += g[9]*3.0f*s35_8*(x2-y2);
    gx += g[10]*s105*y*z;                    gy += g[10]*s105*x*z;          gz += g[10]*s105*x*y;
    gy += g[11]*s21_8*(5.0f*z2-1.0f);        gz += g[11]*10.0f*s21_8*y*z;
    gz += g[12]*0.5f*s7*(15.0f*z2-3.0f);
    gx += g[13]*s21_8*(5.0f*z2-1.0f);        gz += g[13]*10.0f*s21_8*x*z;
    gx += g[14]*s105*x*z;                    gy -= g[14]*s105*y*z;          gz += g[14]*0.5f*s105*(x2-y2);
    gx += g[15]*3.0f*s35_8*(x2-y2);          gy -= g[15]*6.0f*s35_8*x*y;
    // radial derivative
    float u = r/RMAX;
    float u2=u*u, u4=u2*u2, u5=u4*u, u6=u5*u;
    float env  = (u<1.0f) ? (1.0f - 21.0f*u5 + 35.0f*u6 - 15.0f*u6*u) : 0.0f;
    float denv = (u<1.0f) ? ((-105.0f*u4 + 210.0f*u5 - 105.0f*u6)/RMAX) : 0.0f;
    const float c0 = 0.6324555320336759f;
    const float* ge = g_ef + (size_t)e*NB;
    float gr = 0.0f;
#pragma unroll
    for (int nb = 1; nb <= NB; ++nb) {
        float w = (float)nb * (float)M_PI / RMAX;
        float sn = sinf(w*r), cs = cosf(w*r);
        float bess = c0*sn*inv;
        float dbess = c0*(w*cs*inv - sn*inv*inv);
        gr += ge[nb-1]*(dbess*env + bess*denv);
    }
    float dot = gx*x + gy*y + gz*z;
    float gvx = (gx - dot*x)*inv + gr*x;
    float gvy = (gy - dot*y)*inv + gr*y;
    float gvz = (gz - dot*z)*inv + gr*z;
    int s = ei[e], rc = ei[E+e];
    // forces = -dE/dpos ; dE/dpos[recv] += gvec, dE/dpos[send] -= gvec
    atomicAdd(&forces[rc*3+0], -gvx); atomicAdd(&forces[rc*3+1], -gvy); atomicAdd(&forces[rc*3+2], -gvz);
    atomicAdd(&forces[s*3+0],  gvx); atomicAdd(&forces[s*3+1],  gvy); atomicAdd(&forces[s*3+2],  gvz);
}

// ---------------- host driver ----------------
static inline int ceil_div(int a, int b){ return (a+b-1)/b; }

extern "C" void kernel_launch(void* const* d_in, const int* in_sizes, int n_in,
                              void* d_out, int out_size, void* d_ws, size_t ws_size,
                              hipStream_t stream)
{
    (void)in_sizes; (void)n_in; (void)out_size; (void)ws_size;
    const float* positions = (const float*)d_in[0];
    const float* x         = (const float*)d_in[1];
    const float* shifts    = (const float*)d_in[2];
    const float* W_embed   = (const float*)d_in[3];
    const float* W_up      = (const float*)d_in[4];
    const float* Wm1       = (const float*)d_in[5];
    const float* Wm2       = (const float*)d_in[6];
    const float* Wm3       = (const float*)d_in[7];
    const float* W_out     = (const float*)d_in[8];
    const float* W_sc      = (const float*)d_in[9];
    const float* a_sc      = (const float*)d_in[10];
    const float* w_poly    = (const float*)d_in[11];
    const float* W_prod    = (const float*)d_in[12];
    const float* w_r0      = (const float*)d_in[13];
    const float* W_r1      = (const float*)d_in[14];
    const float* w_r2      = (const float*)d_in[15];
    const int*   edge_idx  = (const int*)d_in[16];
    const int*   batch     = (const int*)d_in[17];

    float* out_snap   = (float*)d_out;
    float* out_node_e = out_snap + NG;
    float* out_forces = out_node_e + N;

    // workspace carve
    size_t off = 0;
    auto alloc = [&](size_t bytes) -> char* {
        off = (off + 255) & ~(size_t)255;
        char* p = (char*)d_ws + off;
        off += bytes;
        return p;
    };
    const int CC = C*C;              // 4096
    const int NMC = N*M*C;           // 4,096,000
    // f16 weights (transposed + plain)
    _Float16* hW_upT   = (_Float16*)alloc(sizeof(_Float16)*NL*4*CC);
    _Float16* hW_up    = (_Float16*)alloc(sizeof(_Float16)*NL*4*CC);
    _Float16* hW_outT  = (_Float16*)alloc(sizeof(_Float16)*NL*4*CC);
    _Float16* hW_out   = (_Float16*)alloc(sizeof(_Float16)*NL*4*CC);
    _Float16* hW_scT   = (_Float16*)alloc(sizeof(_Float16)*NL*4*CC);
    _Float16* hW_sc    = (_Float16*)alloc(sizeof(_Float16)*NL*4*CC);
    _Float16* hW_prodT = (_Float16*)alloc(sizeof(_Float16)*NL*4*CC);
    _Float16* hW_prod  = (_Float16*)alloc(sizeof(_Float16)*NL*4*CC);
    _Float16* hWm2T    = (_Float16*)alloc(sizeof(_Float16)*NL*CC);
    _Float16* hWm2     = (_Float16*)alloc(sizeof(_Float16)*NL*CC);
    _Float16* hWm3T    = (_Float16*)alloc(sizeof(_Float16)*NL*C*256);
    _Float16* hWm3     = (_Float16*)alloc(sizeof(_Float16)*NL*C*256);
    _Float16* h16      = (_Float16*)alloc(sizeof(_Float16)*(size_t)E*256); // shared f16 staging

    float* vecb = (float*)alloc(sizeof(float)*E*3);
    float* rlen = (float*)alloc(sizeof(float)*E);
    float* ea   = (float*)alloc(sizeof(float)*E*16);
    float* ef   = (float*)alloc(sizeof(float)*E*NB);
    float* nf[NL+1];
    for (int i = 0; i <= NL; ++i) nf[i] = (float*)alloc(sizeof(float)*NMC);
    float* asc[NL];
    asc[0] = (float*)alloc(sizeof(float)*N*C);
    asc[1] = (float*)alloc(sizeof(float)*N*C);
    float *h0[NL], *z1[NL], *z2[NL], *w3b[NL], *agg[NL], *outb[NL], *gateb[NL];
    for (int i = 0; i < NL; ++i) {
        h0[i]   = (float*)alloc(sizeof(float)*N*C);
        z1[i]   = (float*)alloc(sizeof(float)*E*C);
        z2[i]   = (float*)alloc(sizeof(float)*E*C);
        w3b[i]  = (float*)alloc(sizeof(float)*(size_t)E*256);
        agg[i]  = (float*)alloc(sizeof(float)*NMC);
        outb[i] = (float*)alloc(sizeof(float)*NMC);
        gateb[i]= (float*)alloc(sizeof(float)*N*C);
    }
    float* g_nf2 = (float*)alloc(sizeof(float)*NMC);
    float* g_nf1 = (float*)alloc(sizeof(float)*NMC);
    float* g_og  = (float*)alloc(sizeof(float)*NMC);
    float* g_agg = (float*)alloc(sizeof(float)*NMC);
    float* g_w3  = (float*)alloc(sizeof(float)*(size_t)E*256);
    float* g_z2  = (float*)alloc(sizeof(float)*E*C);
    float* g_z1  = (float*)alloc(sizeof(float)*E*C);
    float* g_h0  = (float*)alloc(sizeof(float)*N*C);
    float* g_ea  = (float*)alloc(sizeof(float)*E*16);
    float* g_ef  = (float*)alloc(sizeof(float)*E*NB);

    const int TB = 256;
    #define GRID1(n) dim3(ceil_div((n), TB)), dim3(TB)

    // ---- weight prep ----
    bct_k<<<GRID1(NL*4*CC), 0, stream>>>(W_up,   hW_upT,   NL*4, C, C);
    cvt_k<<<GRID1(NL*4*CC), 0, stream>>>(W_up,   hW_up,    NL*4*CC);
    bct_k<<<GRID1(NL*4*CC), 0, stream>>>(W_out,  hW_outT,  NL*4, C, C);
    cvt_k<<<GRID1(NL*4*CC), 0, stream>>>(W_out,  hW_out,   NL*4*CC);
    bct_k<<<GRID1(NL*4*CC), 0, stream>>>(W_sc,   hW_scT,   NL*4, C, C);
    cvt_k<<<GRID1(NL*4*CC), 0, stream>>>(W_sc,   hW_sc,    NL*4*CC);
    bct_k<<<GRID1(NL*4*CC), 0, stream>>>(W_prod, hW_prodT, NL*4, C, C);
    cvt_k<<<GRID1(NL*4*CC), 0, stream>>>(W_prod, hW_prod,  NL*4*CC);
    bct_k<<<GRID1(NL*CC), 0, stream>>>(Wm2, hWm2T, NL, C, C);
    cvt_k<<<GRID1(NL*CC), 0, stream>>>(Wm2, hWm2,  NL*CC);
    bct_k<<<GRID1(NL*C*256), 0, stream>>>(Wm3, hWm3T, NL, C, 256);
    cvt_k<<<GRID1(NL*C*256), 0, stream>>>(Wm3, hWm3,  NL*C*256);

    // ---- edge geometry + embedding ----
    edge_geom_k<<<GRID1(E), 0, stream>>>(positions, shifts, edge_idx, vecb, rlen, ea, ef);
    embed_k<<<GRID1(NMC), 0, stream>>>(x, W_embed, nf[0]);
    asc_k<<<GRID1(N*C), 0, stream>>>(x, a_sc, asc[0], asc[1]);

    // ---- forward layers ----
    for (int i = 0; i < NL; ++i) {
        // nf -> f16 ; h0 = nf[:,0,:] @ W_up[i,0] ; sc -> nf[i+1]
        cvt_k<<<GRID1(NMC), 0, stream>>>(nf[i], h16, NMC);
        gemm_k<<<dim3(NTILE, C/16), dim3(128), 0, stream>>>(h16, M*C,
            hW_upT + (size_t)i*4*CC, C, h0[i], C, C, N, 0);
        perl_k<<<dim3(NTILE, M), dim3(128), 0, stream>>>(h16,
            hW_scT + (size_t)i*4*CC, nf[i+1], 0);
        scale_asc_k<<<GRID1(NMC), 0, stream>>>(nf[i+1], asc[i]);
        // edge MLP
        mlp1_k<<<GRID1(E*C), 0, stream>>>(ef, Wm1 + (size_t)i*NB*C, z1[i]);
        cvt_silu_k<<<GRID1(E*C), 0, stream>>>(z1[i], h16, E*C);
        gemm_k<<<dim3(ETILE, C/16), dim3(128), 0, stream>>>(h16, C,
            hWm2T + (size_t)i*CC, C, z2[i], C, C, E, 0);
        cvt_silu_k<<<GRID1(E*C), 0, stream>>>(z2[i], h16, E*C);
        gemm_k<<<dim3(ETILE, 256/16), dim3(128), 0, stream>>>(h16, C,
            hWm3T + (size_t)i*C*256, C, w3b[i], 256, C, E, 0);
        // message + aggregate
        zero_k<<<GRID1(NMC), 0, stream>>>(agg[i], NMC);
        msg_k<<<dim3(E), dim3(C), 0, stream>>>(w3b[i], ea, h0[i], edge_idx, agg[i]);
        // out = per_l(agg, W_out)
        cvt_k<<<GRID1(NMC), 0, stream>>>(agg[i], h16, NMC);
        perl_k<<<dim3(NTILE, M), dim3(128), 0, stream>>>(h16,
            hW_outT + (size_t)i*4*CC, outb[i], 0);
        // gate, og (f16) ; nf[i+1] += per_l(og, W_prod)
        gate_og_k<<<GRID1(NMC), 0, stream>>>(outb[i], w_poly + (size_t)i*3*C, gateb[i], h16);
        perl_k<<<dim3(NTILE, M), dim3(128), 0, stream>>>(h16,
            hW_prodT + (size_t)i*4*CC, nf[i+1], 1);
        // readout
        if (i == 0) ne0_k<<<GRID1(N), 0, stream>>>(nf[1], w_r0, out_node_e);
        else        ne1_k<<<GRID1(N), 0, stream>>>(nf[2], W_r1, w_r2, out_node_e);
    }
    zero_k<<<GRID1(NG), 0, stream>>>(out_snap, NG);
    snap_k<<<GRID1(N), 0, stream>>>(out_node_e, batch, out_snap);

    // ---- backward ----
    zero_k<<<GRID1(E*16), 0, stream>>>(g_ea, E*16);
    zero_k<<<GRID1(E*NB), 0, stream>>>(g_ef, E*NB);
    zero_k<<<GRID1(N*3), 0, stream>>>(out_forces, N*3);
    zero_k<<<GRID1(NMC), 0, stream>>>(g_nf2, NMC);
    gnf2_init_k<<<GRID1(N), 0, stream>>>(nf[2], W_r1, w_r2, g_nf2);

    float* g_cur = g_nf2;
    for (int i = NL-1; i >= 0; --i) {
        // g_og = per_l(g_cur, W_prod^T)
        cvt_k<<<GRID1(NMC), 0, stream>>>(g_cur, h16, NMC);
        perl_k<<<dim3(NTILE, M), dim3(128), 0, stream>>>(h16,
            hW_prod + (size_t)i*4*CC, g_og, 0);
        if (i > 0) {
            // g_nf_prev = per_l(g_cur * asc, W_sc^T)
            gsc_cvt_k<<<GRID1(NMC), 0, stream>>>(g_cur, asc[i], h16);
            perl_k<<<dim3(NTILE, M), dim3(128), 0, stream>>>(h16,
                hW_sc + (size_t)i*4*CC, g_nf1, 0);
        }
        // gate backward: g_og -> g_out in place
        gate_bwd_k<<<GRID1(N*C), 0, stream>>>(outb[i], gateb[i],
            w_poly + (size_t)i*3*C, g_og);
        // g_agg = per_l(g_out, W_out^T)
        cvt_k<<<GRID1(NMC), 0, stream>>>(g_og, h16, NMC);
        perl_k<<<dim3(NTILE, M), dim3(128), 0, stream>>>(h16,
            hW_out + (size_t)i*4*CC, g_agg, 0);
        // message backward
        zero_k<<<GRID1(N*C), 0, stream>>>(g_h0, N*C);
        msg_bwd_k<<<dim3(E), dim3(C), 0, stream>>>(g_agg, w3b[i], ea, h0[i],
            edge_idx, g_w3, g_h0, g_ea);
        // edge MLP backward
        cvt_k<<<GRID1(E*256), 0, stream>>>(g_w3, h16, E*256);
        gemm_k<<<dim3(ETILE, C/16), dim3(128), 0, stream>>>(h16, 256,
            hWm3 + (size_t)i*C*256, 256, g_z2, C, 256, E, 0);
        dsilu_mul_k<<<GRID1(E*C), 0, stream>>>(g_z2, z2[i], E*C);
        cvt_k<<<GRID1(E*C), 0, stream>>>(g_z2, h16, E*C);
        gemm_k<<<dim3(ETILE, C/16), dim3(128), 0, stream>>>(h16, C,
            hWm2 + (size_t)i*CC, C, g_z1, C, C, E, 0);
        dsilu_mul_k<<<GRID1(E*C), 0, stream>>>(g_z1, z1[i], E*C);
        mlp1_bwd_k<<<GRID1(E), 0, stream>>>(g_z1, Wm1 + (size_t)i*NB*C, g_ef);
        if (i > 0) {
            // g_nf_prev[:, 0, :] += g_h0 @ W_up^T
            cvt_k<<<GRID1(N*C), 0, stream>>>(g_h0, h16, N*C);
            gemm_k<<<dim3(NTILE, C/16), dim3(128), 0, stream>>>(h16, C,
                hW_up + (size_t)i*4*CC, C, g_nf1, M*C, C, N, 1);
            // + readout term from layer 0
            add_wr0_k<<<GRID1(N*C), 0, stream>>>(g_nf1, w_r0);
            g_cur = g_nf1;
        }
    }
    // geometric backward -> forces
    edge_bwd_k<<<GRID1(E), 0, stream>>>(vecb, rlen, g_ea, g_ef, edge_idx, out_forces);
    #undef GRID1
}
